// FoodBankGNN_55903294325076
// MI455X (gfx1250) — matrix operand order repaired and checked by reading, compiled, not run on hardware
//
#include <hip/hip_runtime.h>

// ---------------------------------------------------------------------------
// GCN (3x GCNConv) + edge MLP head for MI455X (gfx1250, wave32, WMMA).
// Heavy math (node GEMMs + 79-GFLOP edge MLP) on v_wmma_f32_16x16x32_bf16
// with f32 accumulate; graph aggregation via global f32 atomics.
// Edge MLP: 2 M-tiles (32 edges) per wave so each B tile feeds 2 WMMAs;
// Wm2 staged to LDS via async global->LDS (ASYNCcnt path).
// ---------------------------------------------------------------------------

typedef __attribute__((ext_vector_type(16))) __bf16 v16bf;
typedef __attribute__((ext_vector_type(8)))  __bf16 v8bf;
typedef __attribute__((ext_vector_type(8)))  float  v8f;
typedef __attribute__((ext_vector_type(4)))  float  v4f;

// Types matching the async-to-LDS builtin signature:
//   (v4i addrspace(1)*, v4i addrspace(3)*, imm offset, imm cpol)
typedef int v4i_g __attribute__((vector_size(16)));
typedef __attribute__((address_space(1))) v4i_g as1_v4i;
typedef __attribute__((address_space(3))) v4i_g as3_v4i;

#define N_NODES 50000
#define N_EDGES 800000
#define D 128

#if __has_builtin(__builtin_amdgcn_global_load_async_to_lds_b128) && \
    __has_builtin(__builtin_amdgcn_s_wait_asynccnt)
#define HAS_ASYNC_LDS 1
#else
#define HAS_ASYNC_LDS 0
#endif

__device__ __forceinline__ v16bf cat16(v8bf lo, v8bf hi) {
    return __builtin_shufflevector(lo, hi, 0,1,2,3,4,5,6,7,8,9,10,11,12,13,14,15);
}

__device__ __forceinline__ v8f wmma_bf16(v16bf a, v16bf b, v8f c) {
    return __builtin_amdgcn_wmma_f32_16x16x32_bf16(
        /*neg_a=*/false, a, /*neg_b=*/false, b,
        /*c_mod=*/(short)0, c, /*reuse_a=*/false, /*reuse_b=*/false);
}

// Load the A operand (16x32 bf16 tile) for rows starting at `rowptr` (row-major,
// stride D elements). Lane L: M = L&15; lanes 0-15 carry K {0..7,16..23},
// lanes 16-31 carry K {8..15,24..31} of the 32-wide K chunk.
__device__ __forceinline__ v16bf load_a(const __bf16* __restrict__ rowptr, int hi) {
    v8bf a0 = *(const v8bf*)(rowptr + (hi ? 8  : 0));
    v8bf a1 = *(const v8bf*)(rowptr + (hi ? 24 : 16));
    return cat16(a0, a1);
}

// ---------------------------------------------------------------------------
// Pre-pass kernels
// ---------------------------------------------------------------------------

__global__ void k_convert_bf16(const float* __restrict__ x,
                               __bf16* __restrict__ out, int n4) {
    int t = blockIdx.x * blockDim.x + threadIdx.x;
    if (t >= n4) return;
    v4f v = *(const v4f*)(x + t * 4);
    __bf16* o = out + t * 4;
    o[0] = (__bf16)v.x; o[1] = (__bf16)v.y;
    o[2] = (__bf16)v.z; o[3] = (__bf16)v.w;
}

// W[K][N] f32 (row-major)  ->  packed bf16 [N/16][K/32][32 lanes][16 elems]
// lane: n = nt*16 + (lane&15); k = kc*32 + ((e&8)<<1) + ((lane&16)>>1) + (e&7).
__global__ void k_pack_w(const float* __restrict__ W, __bf16* __restrict__ P,
                         int K, int N) {
    int t = blockIdx.x * blockDim.x + threadIdx.x;
    if (t >= K * N) return;
    int e    = t & 15;
    int lane = (t >> 4) & 31;
    int rem  = t >> 9;               // nt*(K/32) + kc
    int kcC  = K >> 5;
    int kc   = rem % kcC;
    int nt   = rem / kcC;
    int k = kc * 32 + ((e & 8) << 1) + ((lane & 16) >> 1) + (e & 7);
    int n = nt * 16 + (lane & 15);
    P[t] = (__bf16)W[k * N + n];
}

__global__ void k_zero_f32(float* __restrict__ p, int n) {
    int t = blockIdx.x * blockDim.x + threadIdx.x;
    if (t < n) p[t] = 0.0f;
}

__global__ void k_deg(const long long* __restrict__ ei, float* __restrict__ deg) {
    int e = blockIdx.x * blockDim.x + threadIdx.x;
    if (e >= N_EDGES) return;
    atomicAdd(&deg[(int)ei[N_EDGES + e]], 1.0f);
}

__global__ void k_dinv(const float* __restrict__ deg, float* __restrict__ dinv) {
    int i = blockIdx.x * blockDim.x + threadIdx.x;
    if (i >= N_NODES) return;
    dinv[i] = rsqrtf(deg[i] + 1.0f);
}

// ---------------------------------------------------------------------------
// Node GEMM: H[M,128] = Xbf[M,128] @ W(packed).  One wave = 16 rows x 128 cols.
// ---------------------------------------------------------------------------
__global__ void __launch_bounds__(256)
k_gemm_node(const __bf16* __restrict__ X, const __bf16* __restrict__ Wp,
            float* __restrict__ H) {
    int strip = blockIdx.x * 8 + (threadIdx.x >> 5);
    if (strip >= N_NODES / 16) return;
    int lane = threadIdx.x & 31;
    int m    = lane & 15;
    int hi   = (lane & 16) ? 1 : 0;
    const __bf16* row = X + (size_t)(strip * 16 + m) * D;

    v8f zero = {};
    v8f acc[8];
#pragma unroll
    for (int i = 0; i < 8; ++i) acc[i] = zero;

#pragma unroll
    for (int kc = 0; kc < 4; ++kc) {
        v16bf a = load_a(row + kc * 32, hi);
#pragma unroll
        for (int nt = 0; nt < 8; ++nt) {
            v16bf b = *(const v16bf*)(Wp + ((((nt * 4 + kc) * 32) + lane) << 4));
            acc[nt] = wmma_bf16(a, b, acc[nt]);
        }
    }

    int r0 = strip * 16 + hi * 8;
#pragma unroll
    for (int nt = 0; nt < 8; ++nt) {
        int col = nt * 16 + m;
#pragma unroll
        for (int r = 0; r < 8; ++r)
            H[(size_t)(r0 + r) * D + col] = acc[nt][r];
    }
}

// agg[i] = dinv[i]^2 * H[i] + bias   (self-loop term + bias, pre-scatter)
__global__ void k_agg_init(const float* __restrict__ H,
                           const float* __restrict__ dinv,
                           const float* __restrict__ bias,
                           float* __restrict__ A) {
    int t = blockIdx.x * blockDim.x + threadIdx.x;
    if (t >= N_NODES * 32) return;
    int i = t >> 5;
    int c = (t & 31) << 2;
    float s = dinv[i]; s = s * s;
    v4f h = *(const v4f*)(H + (size_t)i * D + c);
    v4f b = *(const v4f*)(bias + c);
    v4f o;
    o.x = h.x * s + b.x; o.y = h.y * s + b.y;
    o.z = h.z * s + b.z; o.w = h.w * s + b.w;
    *(v4f*)(A + (size_t)i * D + c) = o;
}

// agg[dst] += dinv[src]*dinv[dst] * H[src]   (one thread = one edge x 4 cols)
__global__ void k_scatter(const float* __restrict__ H,
                          const float* __restrict__ dinv,
                          const long long* __restrict__ ei,
                          float* __restrict__ A) {
    int t = blockIdx.x * blockDim.x + threadIdx.x;
    if (t >= N_EDGES * 32) return;
    int e = t >> 5;
    int c = (t & 31) << 2;
    int s = (int)ei[e];
    int d = (int)ei[N_EDGES + e];
    float nrm = dinv[s] * dinv[d];
    v4f h = *(const v4f*)(H + (size_t)s * D + c);
    float* o = A + (size_t)d * D + c;
    atomicAdd(o + 0, h.x * nrm);
    atomicAdd(o + 1, h.y * nrm);
    atomicAdd(o + 2, h.z * nrm);
    atomicAdd(o + 3, h.w * nrm);
}

// out_bf16 = bf16( relu?(agg) )
__global__ void k_act_cvt(const float* __restrict__ A, __bf16* __restrict__ O,
                          int doRelu) {
    int t = blockIdx.x * blockDim.x + threadIdx.x;
    if (t >= N_NODES * 32) return;
    int i = t >> 5;
    int c = (t & 31) << 2;
    v4f v = *(const v4f*)(A + (size_t)i * D + c);
    if (doRelu) {
        v.x = fmaxf(v.x, 0.f); v.y = fmaxf(v.y, 0.f);
        v.z = fmaxf(v.z, 0.f); v.w = fmaxf(v.w, 0.f);
    }
    __bf16* o = O + (size_t)i * D + c;
    o[0] = (__bf16)v.x; o[1] = (__bf16)v.y;
    o[2] = (__bf16)v.z; o[3] = (__bf16)v.w;
}

// ---------------------------------------------------------------------------
// Edge MLP head: es = relu([h3[s] | h3[d]] @ Wm1 + bm1) @ Wm2 + bm2
// One wave = 2 x 16-edge tiles (32 edges): each B tile feeds two WMMAs.
// GEMM1: K=256 streamed from gathered h3 rows; ReLU -> LDS bf16 re-layout;
// GEMM2: K=128 with Wm2 staged in LDS (async global->LDS, ASYNCcnt).
// 192 WMMAs per wave. Block = 4 waves (128 threads), 64 KB static LDS.
// ---------------------------------------------------------------------------
__global__ void __launch_bounds__(128)
k_edge_mlp(const __bf16* __restrict__ H3,
           const __bf16* __restrict__ Wp1, const __bf16* __restrict__ Wp2,
           const float* __restrict__ bm1,  const float* __restrict__ bm2,
           const long long* __restrict__ ei, float* __restrict__ out) {
    __shared__ __bf16 hid[4][2][16 * D];   // 8 KB per wave hidden tiles
    __shared__ __bf16 wm2s[D * D];         // 32 KB packed Wm2

    const int tid  = threadIdx.x;
    const int w    = tid >> 5;
    const int lane = tid & 31;
    const int m    = lane & 15;
    const int hi   = (lane & 16) ? 1 : 0;

    // ---- Stage packed Wm2 into LDS (overlapped with GEMM1 when async) ----
    // 32 KB = 2048 x 16B chunks, 16 chunks per thread.
#if HAS_ASYNC_LDS
#pragma unroll
    for (int i = 0; i < 16; ++i) {
        int off = (tid + i * 128) * 8;     // bf16 elements (16B chunks)
        __builtin_amdgcn_global_load_async_to_lds_b128(
            (as1_v4i*)(Wp2 + off), (as3_v4i*)(wm2s + off), 0, 0);
    }
#else
#pragma unroll
    for (int i = 0; i < 16; ++i) {
        int off = (tid + i * 128) * 8;
        *(v8bf*)(wm2s + off) = *(const v8bf*)(Wp2 + off);
    }
#endif

    // Two 16-edge tiles per wave.
    const int tile0 = (blockIdx.x * 4 + w) * 2;
    const int e0 = tile0 * 16 + m;
    const int e1 = e0 + 16;
    const int s0 = (int)ei[e0], d0 = (int)ei[N_EDGES + e0];
    const int s1 = (int)ei[e1], d1 = (int)ei[N_EDGES + e1];
    const __bf16* rs0 = H3 + (size_t)s0 * D;
    const __bf16* rd0 = H3 + (size_t)d0 * D;
    const __bf16* rs1 = H3 + (size_t)s1 * D;
    const __bf16* rd1 = H3 + (size_t)d1 * D;

    v8f zero = {};
    v8f acc0[8], acc1[8];
#pragma unroll
    for (int i = 0; i < 8; ++i) { acc0[i] = zero; acc1[i] = zero; }

    // ---- GEMM1: ef(32x256) @ Wm1(256x128), B shared across both M tiles ----
#pragma unroll
    for (int kc = 0; kc < 8; ++kc) {
        const __bf16* b0 = (kc < 4) ? (rs0 + kc * 32) : (rd0 + (kc - 4) * 32);
        const __bf16* b1 = (kc < 4) ? (rs1 + kc * 32) : (rd1 + (kc - 4) * 32);
        v16bf a0 = load_a(b0, hi);
        v16bf a1 = load_a(b1, hi);
#pragma unroll
        for (int nt = 0; nt < 8; ++nt) {
            v16bf b = *(const v16bf*)(Wp1 + ((((nt * 8 + kc) * 32) + lane) << 4));
            acc0[nt] = wmma_bf16(a0, b, acc0[nt]);
            acc1[nt] = wmma_bf16(a1, b, acc1[nt]);
        }
    }

    // bias + ReLU -> bf16 hidden tiles in LDS (A-operand source for GEMM2)
#pragma unroll
    for (int nt = 0; nt < 8; ++nt) {
        int col = nt * 16 + m;
        float bb = bm1[col];
#pragma unroll
        for (int r = 0; r < 8; ++r) {
            int row = r + hi * 8;
            hid[w][0][row * D + col] = (__bf16)fmaxf(acc0[nt][r] + bb, 0.0f);
            hid[w][1][row * D + col] = (__bf16)fmaxf(acc1[nt][r] + bb, 0.0f);
        }
        acc0[nt] = zero;
        acc1[nt] = zero;
    }
#if HAS_ASYNC_LDS
    __builtin_amdgcn_s_wait_asynccnt(0);
#endif
    __syncthreads();   // hidden tiles + staged Wm2 visible block-wide

    // ---- GEMM2: hidden(32x128) @ Wm2(128x128), B from LDS ----
#pragma unroll
    for (int kc = 0; kc < 4; ++kc) {
        v16bf a0 = load_a(&hid[w][0][m * D + kc * 32], hi);
        v16bf a1 = load_a(&hid[w][1][m * D + kc * 32], hi);
#pragma unroll
        for (int nt = 0; nt < 8; ++nt) {
            v16bf b = *(const v16bf*)(wm2s + ((((nt * 4 + kc) * 32) + lane) << 4));
            acc0[nt] = wmma_bf16(a0, b, acc0[nt]);
            acc1[nt] = wmma_bf16(a1, b, acc1[nt]);
        }
    }

    int r0 = tile0 * 16 + hi * 8;
#pragma unroll
    for (int nt = 0; nt < 8; ++nt) {
        int col = nt * 16 + m;
        float bb = bm2[col];
#pragma unroll
        for (int r = 0; r < 8; ++r) {
            out[(size_t)(r0 + r) * D + col]      = acc0[nt][r] + bb;
            out[(size_t)(r0 + 16 + r) * D + col] = acc1[nt][r] + bb;
        }
    }
}

// ---------------------------------------------------------------------------
// Host launcher
// ---------------------------------------------------------------------------
extern "C" void kernel_launch(void* const* d_in, const int* in_sizes, int n_in,
                              void* d_out, int out_size, void* d_ws, size_t ws_size,
                              hipStream_t stream) {
    const float*     x   = (const float*)d_in[0];
    const long long* ei  = (const long long*)d_in[1];   // int64 [2, E]
    const float* W1 = (const float*)d_in[2];  const float* b1  = (const float*)d_in[3];
    const float* W2 = (const float*)d_in[4];  const float* b2  = (const float*)d_in[5];
    const float* W3 = (const float*)d_in[6];  const float* b3  = (const float*)d_in[7];
    const float* Wm1= (const float*)d_in[8];  const float* bm1 = (const float*)d_in[9];
    const float* Wm2= (const float*)d_in[10]; const float* bm2 = (const float*)d_in[11];
    float* out = (float*)d_out;

    // Workspace carve-up (256B aligned), ~77.5 MB total.
    char* ws = (char*)d_ws;
    size_t off = 0;
    auto alloc = [&](size_t bytes) -> void* {
        off = (off + 255) & ~(size_t)255;
        void* p = ws + off;
        off += bytes;
        return p;
    };
    __bf16* xbfA = (__bf16*)alloc((size_t)N_NODES * D * 2);
    __bf16* xbfB = (__bf16*)alloc((size_t)N_NODES * D * 2);
    float*  hW   = (float*) alloc((size_t)N_NODES * D * 4);
    float*  agg  = (float*) alloc((size_t)N_NODES * D * 4);
    float*  deg  = (float*) alloc((size_t)N_NODES * 4);
    float*  dinv = (float*) alloc((size_t)N_NODES * 4);
    __bf16* Wp1  = (__bf16*)alloc(128 * 128 * 2);
    __bf16* Wp2  = (__bf16*)alloc(128 * 128 * 2);
    __bf16* Wp3  = (__bf16*)alloc(128 * 128 * 2);
    __bf16* Wpm1 = (__bf16*)alloc(256 * 128 * 2);
    __bf16* Wpm2 = (__bf16*)alloc(128 * 128 * 2);
    (void)ws_size; (void)in_sizes; (void)n_in; (void)out_size;

    const int TB = 256;
    const int nodeVec = N_NODES * 32;              // node x 4-col groups
    const int edgeVec = N_EDGES * 32;

    // Pre-passes
    k_convert_bf16<<<(N_NODES * D / 4 + TB - 1) / TB, TB, 0, stream>>>(x, xbfA, N_NODES * D / 4);
    k_pack_w<<<(128 * 128 + TB - 1) / TB, TB, 0, stream>>>(W1,  Wp1,  128, 128);
    k_pack_w<<<(128 * 128 + TB - 1) / TB, TB, 0, stream>>>(W2,  Wp2,  128, 128);
    k_pack_w<<<(128 * 128 + TB - 1) / TB, TB, 0, stream>>>(W3,  Wp3,  128, 128);
    k_pack_w<<<(256 * 128 + TB - 1) / TB, TB, 0, stream>>>(Wm1, Wpm1, 256, 128);
    k_pack_w<<<(128 * 128 + TB - 1) / TB, TB, 0, stream>>>(Wm2, Wpm2, 128, 128);

    // Degrees -> dinv
    k_zero_f32<<<(N_NODES + TB - 1) / TB, TB, 0, stream>>>(deg, N_NODES);
    k_deg<<<N_EDGES / TB, TB, 0, stream>>>(ei, deg);
    k_dinv<<<(N_NODES + TB - 1) / TB, TB, 0, stream>>>(deg, dinv);

    const int gemmBlocks = (N_NODES / 16 + 7) / 8;  // 8 waves per block

    // Layer 1: xbfA -> xbfB (ReLU)
    k_gemm_node<<<gemmBlocks, TB, 0, stream>>>(xbfA, Wp1, hW);
    k_agg_init<<<(nodeVec + TB - 1) / TB, TB, 0, stream>>>(hW, dinv, b1, agg);
    k_scatter<<<(edgeVec + TB - 1) / TB, TB, 0, stream>>>(hW, dinv, ei, agg);
    k_act_cvt<<<(nodeVec + TB - 1) / TB, TB, 0, stream>>>(agg, xbfB, 1);

    // Layer 2: xbfB -> xbfA (ReLU)
    k_gemm_node<<<gemmBlocks, TB, 0, stream>>>(xbfB, Wp2, hW);
    k_agg_init<<<(nodeVec + TB - 1) / TB, TB, 0, stream>>>(hW, dinv, b2, agg);
    k_scatter<<<(edgeVec + TB - 1) / TB, TB, 0, stream>>>(hW, dinv, ei, agg);
    k_act_cvt<<<(nodeVec + TB - 1) / TB, TB, 0, stream>>>(agg, xbfA, 1);

    // Layer 3: xbfA -> xbfB (no ReLU) == h3
    k_gemm_node<<<gemmBlocks, TB, 0, stream>>>(xbfA, Wp3, hW);
    k_agg_init<<<(nodeVec + TB - 1) / TB, TB, 0, stream>>>(hW, dinv, b3, agg);
    k_scatter<<<(edgeVec + TB - 1) / TB, TB, 0, stream>>>(hW, dinv, ei, agg);
    k_act_cvt<<<(nodeVec + TB - 1) / TB, TB, 0, stream>>>(agg, xbfB, 0);

    // Edge MLP head -> d_out [E,128] f32
    // 50000 tiles / (4 waves * 2 tiles) = 6250 blocks of 128 threads.
    k_edge_mlp<<<N_EDGES / 16 / 8, 128, 0, stream>>>(xbfB, Wpm1, Wpm2, bm1, bm2, ei, out);
}
